// SingleHead_21105469292900
// MI455X (gfx1250) — matrix operand order repaired and checked by reading, not compiled, not run on hardware
//
#include <hip/hip_runtime.h>
#include <hip/hip_bf16.h>

// Problem constants (from reference): B=8, T=2048, D_MODEL=1024, D_HEAD=64
#define BB   8
#define TT   2048
#define CM   1024
#define DH   64

typedef __attribute__((ext_vector_type(16))) __bf16 v16bf;
typedef __attribute__((ext_vector_type(8)))  float  v8f;

union FragBF16 {
    v16bf  v;
    __bf16 h[16];
};

// A-operand pair base K for VGPR v in this file's contiguous k order (B uses the same; pairs k,k+1 contiguous):
// lanes 0-15:  V0..V7 -> K=0..15
// lanes 16-31: V0..V7 -> K=16..31
__device__ __forceinline__ int a_pair_k(int v, int hi) {
    return (2 * v) + (hi ? 16 : 0);
}

// ---------------------------------------------------------------------------
// Kernel 1: fused QKV projection, fp32 -> bf16 WMMA -> bf16 Q/K (row-major)
// and V^T ([b][d][t]) in d_ws. 4 waves per block; each wave computes one
// 16-row tile of all of Q,K,V. Weight slab (32 x 192) staged to LDS as bf16
// in [col][k] layout per K step so B-fragments read as packed ds_load_b128.
// ---------------------------------------------------------------------------
__global__ void __launch_bounds__(128)
qkv_proj_kernel(const float* __restrict__ x,
                const float* __restrict__ Wq,
                const float* __restrict__ Wk,
                const float* __restrict__ Wv,
                __bf16* __restrict__ Qb,
                __bf16* __restrict__ Kb,
                __bf16* __restrict__ Vt)
{
    __shared__ __bf16 wlds[192 * 32];   // [col 0..191][k 0..31], 12 KB

    const int tid  = threadIdx.x;
    const int lane = tid & 31;
    const int wave = tid >> 5;
    const int hi   = lane >> 4;         // lane group (0: lanes 0-15, 1: 16-31)
    const int ln   = lane & 15;

    const int  mtile = blockIdx.x * 4 + wave;     // 0..1023
    const long row0  = (long)mtile * 16;          // row in [0, B*T)

    v8f acc[3][4];
    #pragma unroll
    for (int m = 0; m < 3; ++m)
        #pragma unroll
        for (int nt = 0; nt < 4; ++nt)
            acc[m][nt] = v8f{};

    const float* wsrc[3] = { Wq, Wk, Wv };

    for (int kk = 0; kk < CM; kk += 32) {
        // ---- cooperative weight staging: wlds[c*32+k] = (bf16)W[(kk+k)*64 + c%64]
        for (int i = tid; i < 192 * 32; i += 128) {
            const int c = i >> 5;
            const int k = i & 31;
            const float* Wm = wsrc[c >> 6];
            wlds[i] = (__bf16)Wm[(kk + k) * DH + (c & 63)];
        }
        __syncthreads();

        // ---- A fragment: x rows, row m = lane&15, fp32 -> bf16
        FragBF16 a;
        const float* xrow = x + (row0 + ln) * (long)CM + kk;
        #pragma unroll
        for (int v = 0; v < 8; ++v) {
            const int k0 = a_pair_k(v, hi);
            a.h[2 * v]     = (__bf16)xrow[k0];
            a.h[2 * v + 1] = (__bf16)xrow[k0 + 1];
        }
        __builtin_prefetch(xrow + 32, 0, 1);   // next K slab -> global_prefetch_b8

        // ---- 12 WMMAs: 3 matrices x 4 N tiles
        #pragma unroll
        for (int mat = 0; mat < 3; ++mat) {
            #pragma unroll
            for (int nt = 0; nt < 4; ++nt) {
                FragBF16 bfr;
                const __bf16* wp = &wlds[(mat * 64 + nt * 16 + ln) * 32 + (hi ? 16 : 0)];
                #pragma unroll
                for (int v = 0; v < 8; ++v) {
                    bfr.h[2 * v]     = wp[2 * v];
                    bfr.h[2 * v + 1] = wp[2 * v + 1];
                }
                acc[mat][nt] = __builtin_amdgcn_wmma_f32_16x16x32_bf16(
                    false, a.v, false, bfr.v, (short)0, acc[mat][nt], false, false);
            }
        }
        __syncthreads();
    }

    // ---- epilogue: Q/K row-major, V transposed [b][d][t]
    #pragma unroll
    for (int nt = 0; nt < 4; ++nt)
        #pragma unroll
        for (int r = 0; r < 8; ++r) {
            const long rowg = row0 + r + 8 * hi;
            Qb[rowg * DH + nt * 16 + ln] = (__bf16)acc[0][nt][r];
            Kb[rowg * DH + nt * 16 + ln] = (__bf16)acc[1][nt][r];
            const long bidx = rowg >> 11;            // / TT
            const long trow = rowg & (TT - 1);       // % TT
            Vt[bidx * (long)(DH * TT) + (long)(nt * 16 + ln) * TT + trow] =
                (__bf16)acc[2][nt][r];
        }
}

// ---------------------------------------------------------------------------
// Kernel 2: flash-attention. One wave per 16-query tile; key blocks of 32 so
// P (16x32 bf16) feeds a full K=32 WMMA into the output accumulator.
// V is consumed transposed: the B-fragment K axis (key index) is contiguous,
// so each V fragment is two b128 loads instead of 16 scalar d16 loads.
// ---------------------------------------------------------------------------
__global__ void __launch_bounds__(128)
flash_attn_kernel(const __bf16* __restrict__ Qb,
                  const __bf16* __restrict__ Kb,
                  const __bf16* __restrict__ Vt,
                  float* __restrict__ out)
{
    __shared__ __bf16 plds[4][16 * 32];   // per-wave P staging tile, 4 KB

    const int tid  = threadIdx.x;
    const int lane = tid & 31;
    const int wave = tid >> 5;
    const int hi   = lane >> 4;
    const int ln   = lane & 15;

    const int  b      = blockIdx.y;
    const int  qt     = blockIdx.x * 4 + wave;   // 0..127 query tile
    const long base   = (long)b * TT * DH;
    const long vtbase = (long)b * DH * TT;

    // ---- Q A-fragments (d=64 -> two K=32 steps)
    FragBF16 aQ[2];
    const __bf16* qrow = Qb + base + (long)(qt * 16 + ln) * DH;
    #pragma unroll
    for (int s = 0; s < 2; ++s)
        #pragma unroll
        for (int v = 0; v < 8; ++v) {
            const int k0 = s * 32 + a_pair_k(v, hi);
            aQ[s].h[2 * v]     = qrow[k0];
            aQ[s].h[2 * v + 1] = qrow[k0 + 1];
        }

    v8f   o[4];
    float mrow[8], lrow[8];
    #pragma unroll
    for (int nt = 0; nt < 4; ++nt) o[nt] = v8f{};
    #pragma unroll
    for (int r = 0; r < 8; ++r) { mrow[r] = -3.0e38f; lrow[r] = 0.0f; }

    const float scale = 0.03125f;               // 1/sqrt(D_MODEL) = 1/32
    const int   nblk  = (qt + 2) >> 1;          // causal: key blocks of 32
    __bf16*     pl    = &plds[wave][0];

    for (int j = 0; j < nblk; ++j) {
        const int kbase = j * 32;

        // ---- K B-fragments (B = K^T: n = key row, k = head dim; contiguous in d)
        FragBF16 bK[2][2];
        #pragma unroll
        for (int t = 0; t < 2; ++t) {
            const __bf16* krow = Kb + base + (long)(kbase + t * 16 + ln) * DH;
            #pragma unroll
            for (int s = 0; s < 2; ++s)
                #pragma unroll
                for (int v = 0; v < 8; ++v) {
                    const int d0 = s * 32 + 2 * v + (hi ? 16 : 0);
                    bK[t][s].h[2 * v]     = krow[d0];
                    bK[t][s].h[2 * v + 1] = krow[d0 + 1];
                }
        }
        // prefetch next key block (K rows and V^T columns)
        __builtin_prefetch(Kb + base + (long)(kbase + 32 + ln) * DH, 0, 1);
        __builtin_prefetch(Vt + vtbase + (long)ln * TT + kbase + 32, 0, 1);

        // ---- S = Q K^T  (16 x 32 in two 16x16 accumulators)
        v8f sAcc[2];
        sAcc[0] = v8f{}; sAcc[1] = v8f{};
        #pragma unroll
        for (int t = 0; t < 2; ++t)
            #pragma unroll
            for (int s = 0; s < 2; ++s)
                sAcc[t] = __builtin_amdgcn_wmma_f32_16x16x32_bf16(
                    false, aQ[s].v, false, bK[t][s].v, (short)0, sAcc[t], false, false);

        // ---- causal mask + online softmax (fp32 VALU + lane-group shuffles)
        #pragma unroll
        for (int r = 0; r < 8; ++r) {
            const int rowg = qt * 16 + r + 8 * hi;
            float pv[2];
            float best = -3.0e38f;
            #pragma unroll
            for (int t = 0; t < 2; ++t) {
                const int colg = kbase + t * 16 + ln;
                float sv = sAcc[t][r] * scale;
                if (colg > rowg) sv = -3.0e38f;
                pv[t] = sv;
                best  = fmaxf(best, sv);
            }
            #pragma unroll
            for (int m = 1; m < 16; m <<= 1)
                best = fmaxf(best, __shfl_xor(best, m, 16));

            const float mnew  = fmaxf(mrow[r], best);
            const float alpha = __expf(mrow[r] - mnew);
            float rsum = 0.0f;
            #pragma unroll
            for (int t = 0; t < 2; ++t) {
                const float p = __expf(pv[t] - mnew);
                pv[t] = p;
                rsum += p;
            }
            #pragma unroll
            for (int m = 1; m < 16; m <<= 1)
                rsum += __shfl_xor(rsum, m, 16);

            lrow[r] = lrow[r] * alpha + rsum;
            mrow[r] = mnew;
            #pragma unroll
            for (int nt = 0; nt < 4; ++nt) o[nt][r] *= alpha;

            // stage P (C layout -> [m][k] tile); DS ops are in-order per wave
            #pragma unroll
            for (int t = 0; t < 2; ++t)
                pl[(r + 8 * hi) * 32 + t * 16 + ln] = (__bf16)pv[t];
        }

        // ---- reload P as A-fragment (16x32 bf16)
        FragBF16 aP;
        const __bf16* prow = pl + ln * 32;
        #pragma unroll
        for (int v = 0; v < 8; ++v) {
            const int k0 = a_pair_k(v, hi);
            aP.h[2 * v]     = prow[k0];
            aP.h[2 * v + 1] = prow[k0 + 1];
        }

        // ---- O += P V : V^T column (n = out dim across lanes, k = key index
        // contiguous) -> each fragment is 16 consecutive bf16 (two b128 loads)
        #pragma unroll
        for (int nt = 0; nt < 4; ++nt) {
            FragBF16 bV;
            const __bf16* vcol = Vt + vtbase + (long)(nt * 16 + ln) * TT
                                 + kbase + (hi ? 16 : 0);
            #pragma unroll
            for (int v = 0; v < 8; ++v) {
                bV.h[2 * v]     = vcol[2 * v];
                bV.h[2 * v + 1] = vcol[2 * v + 1];
            }
            o[nt] = __builtin_amdgcn_wmma_f32_16x16x32_bf16(
                false, aP.v, false, bV.v, (short)0, o[nt], false, false);
        }
    }

    // ---- epilogue: normalize and store fp32
    #pragma unroll
    for (int nt = 0; nt < 4; ++nt)
        #pragma unroll
        for (int r = 0; r < 8; ++r) {
            const long rowg = qt * 16 + r + 8 * hi;
            out[base + rowg * DH + nt * 16 + ln] = o[nt][r] / lrow[r];
        }
}

// ---------------------------------------------------------------------------
extern "C" void kernel_launch(void* const* d_in, const int* in_sizes, int n_in,
                              void* d_out, int out_size, void* d_ws, size_t ws_size,
                              hipStream_t stream) {
    const float* x  = (const float*)d_in[0];
    const float* Wq = (const float*)d_in[1];
    const float* Wk = (const float*)d_in[2];
    const float* Wv = (const float*)d_in[3];
    float* out = (float*)d_out;

    // d_ws: bf16 Q | K | V^T, each B*T*DH elements (2 MB each, 6 MB total)
    const size_t elems = (size_t)BB * TT * DH;
    __bf16* Qb = (__bf16*)d_ws;
    __bf16* Kb = Qb + elems;
    __bf16* Vt = Kb + elems;

    // (B*T)/16 = 1024 M-tiles, 4 waves per block
    qkv_proj_kernel<<<dim3(256), dim3(128), 0, stream>>>(x, Wq, Wk, Wv, Qb, Kb, Vt);

    // 128 query tiles per batch, 4 waves per block, grid.y = batch
    flash_attn_kernel<<<dim3(32, 8), dim3(128), 0, stream>>>(Qb, Kb, Vt, out);
}
